// NonLocalBlock_31138512896696
// MI455X (gfx1250) — matrix operand order repaired and checked by reading, not compiled, not run on hardware
//
#include <hip/hip_runtime.h>
#include <stdint.h>

#define B_  8
#define N_  4096
#define C_  256
#define CB_ 128
#define M_  2048

#define PLD  136   // LDS row stride (bf16 elems) for phi/g tiles (128 + 8 pad = 272B)
#define SSTR 40    // LDS row stride for per-wave P staging (32 + 8 pad = 80B, 16B-aligned rows)
#define ATW  4     // waves per attention workgroup
#define MT_  (M_ / 32)                         // 64 key tiles

typedef __attribute__((ext_vector_type(16))) __bf16 v16bf;
typedef __attribute__((ext_vector_type(8)))  float  v8f;
typedef unsigned int v4u __attribute__((ext_vector_type(4)));
typedef int          v8i __attribute__((ext_vector_type(8)));
typedef int          v4i __attribute__((ext_vector_type(4)));

union FragU { uint4 q[2]; v16bf v; };

static __device__ __forceinline__ __bf16 u16_as_bf16(unsigned short u) {
  union { unsigned short s; __bf16 b; } v; v.s = u; return v.b;
}
static __device__ __forceinline__ unsigned short f32_to_bf16(float f) {
  unsigned int u = __float_as_uint(f);
  u += 0x7FFFu + ((u >> 16) & 1u);           // round-to-nearest-even
  return (unsigned short)(u >> 16);
}

// A-fragment (16x32 bf16): elems 0..7 <-> K = kb+half*8..+7, elems 8..15 <-> K = kb+16+half*8..+7
static __device__ __forceinline__ v16bf load_a_frag(const unsigned short* rowp, int kb, int half) {
  FragU u;
  u.q[0] = *(const uint4*)(rowp + kb + half * 8);
  u.q[1] = *(const uint4*)(rowp + kb + 16 + half * 8);
  return u.v;
}
// B-fragment (32x16 bf16): lane = N column, elems 0..7 <-> K = half*8..+7, elems 8..15 <-> K = 16+half*8..+7
static __device__ __forceinline__ v16bf load_b_frag16(const unsigned short* p, int ld, int half) {
  v16bf v;
  #pragma unroll
  for (int s = 0; s < 16; ++s) {
    int k = (s < 8) ? (half * 8 + s) : (16 + half * 8 + (s - 8));
    v[s] = u16_as_bf16(p[(size_t)k * ld]);
  }
  return v;
}

// ---------------------------------------------------------------------------
// Tensor Data Mover: DMA a [32 x 128] bf16 tile into LDS with 16B row padding
// (row stride in LDS becomes 272B == PLD*2). D# built per CDNA5 ISA.
// ---------------------------------------------------------------------------
#define HAVE_TDM __has_builtin(__builtin_amdgcn_tensor_load_to_lds)
#if HAVE_TDM
static __device__ __forceinline__ void tdm_load_tile_bf16(unsigned lds_off,
                                                          const unsigned short* gptr) {
  unsigned long long ga = (unsigned long long)(uintptr_t)gptr;
  v4u g0;
  g0[0] = 1u;                                   // count=1 (valid user descriptor)
  g0[1] = lds_off;                              // lds_addr (bytes)
  g0[2] = (unsigned)(ga & 0xFFFFFFFFu);         // global_addr[31:0]
  g0[3] = (unsigned)((ga >> 32) & 0x1FFFFFFu)   // global_addr[56:32]
        | (2u << 30);                           // type = 2 ("image")
  v8i g1;
  g1[0] = (int)((1u << 16)                      // data_size = 2 bytes
              | (1u << 20)                      // pad_enable
              | (5u << 22)                      // pad_interval: 64 DWORDs (256B row)
              | (3u << 25));                    // pad_amount: 4 DWORDs (16B pad)
  g1[1] = (int)((CB_ & 0xFFFF) << 16);          // tensor_dim0 = 128 (bits 79:48)
  g1[2] = (int)((CB_ >> 16) | ((M_ & 0xFFFF) << 16)); // | tensor_dim1 = 2048
  g1[3] = (int)((M_ >> 16) | (CB_ << 16));      // | tile_dim0 = 128
  g1[4] = 32;                                   // tile_dim1 = 32, tile_dim2 = 0
  g1[5] = CB_;                                  // tensor_dim0_stride = 128 (lo32)
  g1[6] = 0;
  g1[7] = 0;
  v4i z = {0, 0, 0, 0};
#if defined(__clang_major__) && (__clang_major__ >= 23)
  v8i z8 = {0, 0, 0, 0, 0, 0, 0, 0};
  __builtin_amdgcn_tensor_load_to_lds(g0, g1, z, z, z8, 0);
#else
  __builtin_amdgcn_tensor_load_to_lds(g0, g1, z, z, 0);
#endif
}
#endif

// ---------------------------------------------------------------------------
// Stage 0: fp32 -> bf16 conversion
// ---------------------------------------------------------------------------
__global__ void cvt_f32_bf16(const float* __restrict__ src,
                             unsigned short* __restrict__ dst, int n) {
  int i = blockIdx.x * blockDim.x + threadIdx.x;
  if (i < n) dst[i] = f32_to_bf16(src[i]);
}

// ---------------------------------------------------------------------------
// Stage 1: projections. wave id -> (p, b, row-tile, col-tile).
// p=0: theta (full, bf16). p=1: phi (maxpool2 rows). p=2: g (maxpool2 rows).
// ---------------------------------------------------------------------------
__global__ void __launch_bounds__(256) proj_kernel(
    const unsigned short* __restrict__ xb,
    const unsigned short* __restrict__ wtb,
    const unsigned short* __restrict__ wpb,
    const unsigned short* __restrict__ wgb,
    unsigned short* __restrict__ theta,
    unsigned short* __restrict__ phiP,
    unsigned short* __restrict__ gP) {
  const int lane = threadIdx.x & 31;
  const int wid  = blockIdx.x * 8 + (threadIdx.x >> 5);
  const int COLT = CB_ / 16;                 // 8
  const int ROWT = N_ / 16;                  // 256

  int ct = wid % COLT;
  int t  = wid / COLT;
  int rt = t % ROWT; t /= ROWT;
  int b  = t % B_;
  int p  = t / B_;                           // 0..2, uniform per wave

  const unsigned short* w = (p == 0) ? wtb : (p == 1) ? wpb : wgb;
  const int half = lane >> 4;
  const int l15  = lane & 15;
  const int rowBase = rt * 16;

  v8f acc;
  #pragma unroll
  for (int e = 0; e < 8; ++e) acc[e] = 0.0f;

  const unsigned short* xrow = xb + ((size_t)b * N_ + rowBase + l15) * C_;
  for (int kb = 0; kb < C_; kb += 32) {
    v16bf a  = load_a_frag(xrow, kb, half);
    v16bf bf = load_b_frag16(w + (size_t)kb * CB_ + ct * 16 + l15, CB_, half);
    acc = __builtin_amdgcn_wmma_f32_16x16x32_bf16(false, a, false, bf,
                                                  (short)0, acc, false, false);
  }

  if (p == 0) {
    #pragma unroll
    for (int r = 0; r < 8; ++r) {
      int row = rowBase + r + 8 * half;
      theta[((size_t)b * N_ + row) * CB_ + ct * 16 + l15] = f32_to_bf16(acc[r]);
    }
  } else {
    unsigned short* dst = (p == 1) ? phiP : gP;
    #pragma unroll
    for (int i = 0; i < 4; ++i) {            // pool adjacent rows (2i, 2i+1)
      float mv = fmaxf(acc[2 * i], acc[2 * i + 1]);
      int prow = rt * 8 + i + 4 * half;
      dst[((size_t)b * M_ + prow) * CB_ + ct * 16 + l15] = f32_to_bf16(mv);
    }
  }
}

// ---------------------------------------------------------------------------
// Stage 2: flash attention (S^T formulation) + output projection + residual.
// Block = 128 threads (4 waves), each wave owns 16 query rows.
// Double-buffered TDM pipeline: wave 0 issues tile t+1's tensor loads before
// waiting for tile t (s_wait_tensorcnt 2; TDM completes in order).
// ---------------------------------------------------------------------------
__global__ void __launch_bounds__(128) attn_kernel(
    const unsigned short* __restrict__ theta,
    const unsigned short* __restrict__ phiP,
    const unsigned short* __restrict__ gP,
    const unsigned short* __restrict__ wob,
    const float* __restrict__ x,
    float* __restrict__ out) {
  __shared__ __align__(16) unsigned short sPhi[2][32 * PLD];
  __shared__ __align__(16) unsigned short sG[2][32 * PLD];
  __shared__ __align__(16) unsigned short sStage[ATW][16 * SSTR];

  const int lane = threadIdx.x & 31;
  const int wv   = threadIdx.x >> 5;
  const int half = lane >> 4;
  const int l15  = lane & 15;

  const int b       = blockIdx.x >> 6;
  const int rtile   = blockIdx.x & 63;
  const int rowBase = rtile * 64 + wv * 16;

  // warm the caches for this wave's theta rows and the shared W_out
  __builtin_prefetch(theta + ((size_t)b * N_ + rowBase + l15) * CB_, 0, 3);
  __builtin_prefetch(wob + (size_t)lane * 2 * C_, 0, 3);

  // theta^T B-fragments (lane = query row n, elem = K), loop invariant
  v16bf tb[4];
  #pragma unroll
  for (int kb = 0; kb < 4; ++kb) {
    #pragma unroll
    tb[kb] = load_a_frag(theta + ((size_t)b * N_ + rowBase + l15) * CB_, kb * 32, half);
  }

  float m1 = -1e30f, s1 = 0.0f;              // per-lane stats for query n = l15
  v8f O[8];
  #pragma unroll
  for (int c = 0; c < 8; ++c) {
    #pragma unroll
    for (int e = 0; e < 8; ++e) O[c][e] = 0.0f;
  }

  unsigned short* stg = &sStage[wv][0];
  const size_t bbase = (size_t)b * M_ * CB_;

#if HAVE_TDM
  const unsigned phiOff[2] = {(unsigned)(size_t)(const void*)&sPhi[0][0],
                              (unsigned)(size_t)(const void*)&sPhi[1][0]};
  const unsigned gOff[2]   = {(unsigned)(size_t)(const void*)&sG[0][0],
                              (unsigned)(size_t)(const void*)&sG[1][0]};
  if (wv == 0) {                             // prologue: DMA tile 0 into buffer 0
    tdm_load_tile_bf16(phiOff[0], phiP + bbase);
    tdm_load_tile_bf16(gOff[0], gP + bbase);
  }
#endif

  for (int t = 0; t < MT_; ++t) {
    int cur = t & 1;
#if HAVE_TDM
    if (wv == 0) {                           // uniform branch; TDM ignores EXEC
      if (t + 1 < MT_) {                     // prefetch next tile into other buffer
        const size_t nb = bbase + (size_t)(t + 1) * 32 * CB_;
        tdm_load_tile_bf16(phiOff[cur ^ 1], phiP + nb);
        tdm_load_tile_bf16(gOff[cur ^ 1], gP + nb);
        __builtin_amdgcn_s_wait_tensorcnt(2);   // tile t done, t+1 in flight
      } else {
        __builtin_amdgcn_s_wait_tensorcnt(0);
      }
    }
    __syncthreads();                         // publish tile t
#else
    cur = 0;
    {
      const size_t base = bbase + (size_t)t * 32 * CB_;
      #pragma unroll
      for (int j = 0; j < 4; ++j) {
        int cidx = threadIdx.x * 4 + j;
        int row  = cidx >> 4;
        int col  = (cidx & 15) * 8;
        *(uint4*)(&sPhi[0][row * PLD + col]) =
            *(const uint4*)(phiP + base + (size_t)row * CB_ + col);
        *(uint4*)(&sG[0][row * PLD + col]) =
            *(const uint4*)(gP + base + (size_t)row * CB_ + col);
      }
    }
    __syncthreads();
#endif
    const unsigned short* Pc = &sPhi[cur][0];
    const unsigned short* Gc = &sG[cur][0];

    // S^T tile (32m x 16n) as two 16x16 C fragments
    v8f T0, T1;
    #pragma unroll
    for (int e = 0; e < 8; ++e) { T0[e] = 0.0f; T1[e] = 0.0f; }
    #pragma unroll
    for (int kb = 0; kb < 4; ++kb) {
      v16bf a0 = load_a_frag(Pc + l15 * PLD, kb * 32, half);
      v16bf a1 = load_a_frag(Pc + (16 + l15) * PLD, kb * 32, half);
      T0 = __builtin_amdgcn_wmma_f32_16x16x32_bf16(false, a0, false, tb[kb],
                                                   (short)0, T0, false, false);
      T1 = __builtin_amdgcn_wmma_f32_16x16x32_bf16(false, a1, false, tb[kb],
                                                   (short)0, T1, false, false);
    }

    // online softmax over m: in-register over 16 values + one cross-half shfl
    float pmax = fmaxf(T0[0], T1[0]);
    #pragma unroll
    for (int r = 1; r < 8; ++r) pmax = fmaxf(pmax, fmaxf(T0[r], T1[r]));
    pmax = fmaxf(pmax, __shfl_xor(pmax, 16, 32));

    float nm  = fmaxf(m1, pmax);
    float sc1 = __expf(m1 - nm);
    m1 = nm;

    float rs = 0.0f;
    #pragma unroll
    for (int r = 0; r < 8; ++r) {            // element (r,lane) = S[n=l15][m = r+8*half (+16)]
      float p0 = __expf(T0[r] - nm);
      float p1 = __expf(T1[r] - nm);
      rs += p0 + p1;
      stg[l15 * SSTR + r + 8 * half]      = f32_to_bf16(p0);
      stg[l15 * SSTR + 16 + r + 8 * half] = f32_to_bf16(p1);
    }
    rs += __shfl_xor(rs, 16, 32);
    s1 = s1 * sc1 + rs;

    #pragma unroll
    for (int r = 0; r < 8; ++r) {            // O row n = r+8*half: fetch that n's scale
      float scO = __shfl(sc1, r + 8 * half, 32);
      #pragma unroll
      for (int c = 0; c < 8; ++c) O[c][r] *= scO;
    }

    asm volatile("s_wait_dscnt 0" ::: "memory");  // P staging visible to own wave

    // P fragment in A layout (16n x 32m): two aligned b128 LDS reads
    v16bf pf = load_a_frag(stg + l15 * SSTR, 0, half);

    // O (16n x 128ch) += P (16n x 32m) @ g_tile (32m x 128ch)
    #pragma unroll
    for (int c = 0; c < 8; ++c) {
      v16bf gb = load_b_frag16(Gc + c * 16 + l15, PLD, half);
      O[c] = __builtin_amdgcn_wmma_f32_16x16x32_bf16(false, pf, false, gb,
                                                     (short)0, O[c], false, false);
    }
    __syncthreads();                         // all waves done reading buffer `cur`
  }

  // epilogue: y = (O / l) @ W_out + x
  float inv1 = 1.0f / s1;                    // per-lane, for query n = l15
  v16bf of[4];
  #pragma unroll
  for (int q = 0; q < 4; ++q) {              // O C-layout -> A-layout, 32 channels at a time
    #pragma unroll
    for (int r = 0; r < 8; ++r) {
      float invO = __shfl(inv1, r + 8 * half, 32);
      int row = r + 8 * half;
      stg[row * SSTR + l15]      = f32_to_bf16(O[2 * q][r] * invO);
      stg[row * SSTR + 16 + l15] = f32_to_bf16(O[2 * q + 1][r] * invO);
    }
    asm volatile("s_wait_dscnt 0" ::: "memory");
    of[q] = load_a_frag(stg + l15 * SSTR, 0, half);
    asm volatile("s_wait_dscnt 0" ::: "memory");
  }

  #pragma unroll 1
  for (int ct = 0; ct < 16; ++ct) {          // 256 output channels / 16
    v8f acc;
    #pragma unroll
    for (int e = 0; e < 8; ++e) acc[e] = 0.0f;
    #pragma unroll
    for (int kb = 0; kb < 4; ++kb) {
      v16bf wf = load_b_frag16(wob + (size_t)(kb * 32) * C_ + ct * 16 + l15, C_, half);
      acc = __builtin_amdgcn_wmma_f32_16x16x32_bf16(false, of[kb], false, wf,
                                                    (short)0, acc, false, false);
    }
    #pragma unroll
    for (int r = 0; r < 8; ++r) {
      int row = rowBase + r + 8 * half;
      size_t idx = ((size_t)b * N_ + row) * C_ + ct * 16 + l15;
      out[idx] = x[idx] + acc[r];
    }
  }
}

// ---------------------------------------------------------------------------
extern "C" void kernel_launch(void* const* d_in, const int* in_sizes, int n_in,
                              void* d_out, int out_size, void* d_ws, size_t ws_size,
                              hipStream_t stream) {
  (void)in_sizes; (void)n_in; (void)out_size; (void)ws_size;
  const float* x  = (const float*)d_in[0];
  const float* Wt = (const float*)d_in[1];
  const float* Wp = (const float*)d_in[2];
  const float* Wg = (const float*)d_in[3];
  const float* Wo = (const float*)d_in[4];
  float* out = (float*)d_out;

  char* ws = (char*)d_ws;
  unsigned short* xb    = (unsigned short*)(ws);             // 16,777,216 B
  unsigned short* theta = (unsigned short*)(ws + 16777216);  //  8,388,608 B
  unsigned short* phiP  = (unsigned short*)(ws + 25165824);  //  4,194,304 B
  unsigned short* gPp   = (unsigned short*)(ws + 29360128);  //  4,194,304 B
  unsigned short* wtb   = (unsigned short*)(ws + 33554432);  //     65,536 B
  unsigned short* wpb   = (unsigned short*)(ws + 33619968);
  unsigned short* wgb   = (unsigned short*)(ws + 33685504);
  unsigned short* wob   = (unsigned short*)(ws + 33751040);  // end 33,816,576 B

  cvt_f32_bf16<<<(B_ * N_ * C_) / 256, 256, 0, stream>>>(x, xb, B_ * N_ * C_);
  cvt_f32_bf16<<<(C_ * CB_) / 256, 256, 0, stream>>>(Wt, wtb, C_ * CB_);
  cvt_f32_bf16<<<(C_ * CB_) / 256, 256, 0, stream>>>(Wp, wpb, C_ * CB_);
  cvt_f32_bf16<<<(C_ * CB_) / 256, 256, 0, stream>>>(Wg, wgb, C_ * CB_);
  cvt_f32_bf16<<<(CB_ * C_) / 256, 256, 0, stream>>>(Wo, wob, CB_ * C_);

  // 3 projections * 8 batches * 256 row-tiles * 8 col-tiles = 49152 waves / 8 per block
  proj_kernel<<<6144, 256, 0, stream>>>(xb, wtb, wpb, wgb, theta, phiP, gPp);

  // 8 batches * 64 row-tiles-of-64 = 512 blocks of 128 threads
  attn_kernel<<<512, 128, 0, stream>>>(theta, phiP, gPp, wob, x, out);
}
